// attention_layer_58935541236541
// MI455X (gfx1250) — matrix operand order repaired and checked by reading, not compiled, not run on hardware
//
#include <hip/hip_runtime.h>
#include <hip/hip_bf16.h>

#define N_TOT 8192
#define IN_C  512
#define OUT_C 256
#define NEG_INF (-1000000000.0f)

typedef __attribute__((ext_vector_type(16))) __bf16 v16bf;
typedef __attribute__((ext_vector_type(8)))  float  v8f;
typedef __attribute__((ext_vector_type(2)))  float  v2f;

// round-to-nearest-even float -> bf16 bits
__device__ __forceinline__ unsigned short f2bf(float f) {
    unsigned int u = __float_as_uint(f);
    unsigned int r = u + 0x7FFFu + ((u >> 16) & 1u);
    return (unsigned short)(r >> 16);
}

// Async global->LDS copy of 16 bytes per lane (CDNA5 VGLOBAL async op,
// tracked with ASYNCcnt). LDS offset = low 32 bits of the generic pointer
// (LDS aperture maps addr[31:0] to the LDS byte offset).
__device__ __forceinline__ void async_copy_b128(void* lds_dst, const void* gsrc) {
    unsigned lds = (unsigned)(size_t)lds_dst;
    asm volatile("global_load_async_to_lds_b128 %0, %1, off"
                 :: "v"(lds), "v"(gsrc)
                 : "memory");
}

// ---------------------------------------------------------------------------
// Kernel 1: h = x @ W_fc + b_fc   (f32 WMMA 16x16x4)
// Writes h (f32, row-major [N, OUT_C]) and hT (bf16, [OUT_C, N] transposed,
// so attention-kernel B fragments are contiguous along K).
// Block: 256 thr (8 waves) computes 16 rows x 256 cols. Wave w -> cols 32w..32w+31.
// ---------------------------------------------------------------------------
__global__ __launch_bounds__(256) void gat_fc_kernel(
    const float* __restrict__ x, const float* __restrict__ Wfc,
    const float* __restrict__ bfc,
    float* __restrict__ h, unsigned short* __restrict__ hT)
{
    const int i0   = blockIdx.x * 16;
    const int tid  = threadIdx.x;
    const int wid  = tid >> 5;
    const int lane = tid & 31;
    const int hi   = lane >> 4;     // which half-wave
    const int m    = lane & 15;
    const int n0   = wid * 32;

    v8f acc0 = {0.f,0.f,0.f,0.f,0.f,0.f,0.f,0.f};
    v8f acc1 = acc0;

    const int row = i0 + m;                       // A-matrix row for this lane
    const float* xr = x + (size_t)row * IN_C + 2 * hi;
    const int na = n0 + m;
    const int nb = n0 + 16 + m;

    #pragma unroll 4
    for (int k = 0; k < IN_C; k += 4) {
        // A frag 16x4 f32: lanes 0-15 K{0,1}, lanes 16-31 K{2,3}
        v2f a = *(const v2f*)(xr + k);
        const int kk = k + 2 * hi;
        // B frag 4x16 f32: lanes 0-15 rows K{0,1}, lanes 16-31 rows K{2,3}
        v2f b0 = { Wfc[(size_t)kk * OUT_C + na], Wfc[(size_t)(kk + 1) * OUT_C + na] };
        v2f b1 = { Wfc[(size_t)kk * OUT_C + nb], Wfc[(size_t)(kk + 1) * OUT_C + nb] };
        acc0 = __builtin_amdgcn_wmma_f32_16x16x4_f32(false, a, false, b0, (short)0, acc0, false, false);
        acc1 = __builtin_amdgcn_wmma_f32_16x16x4_f32(false, a, false, b1, (short)0, acc1, false, false);
    }

    const float bias0 = bfc[na];
    const float bias1 = bfc[nb];
    #pragma unroll
    for (int v = 0; v < 8; ++v) {
        const int r = i0 + v + 8 * hi;            // C/D layout: VGPR v, half hi
        float h0 = acc0[v] + bias0;
        float h1 = acc1[v] + bias1;
        h[(size_t)r * OUT_C + na] = h0;
        h[(size_t)r * OUT_C + nb] = h1;
        hT[(size_t)na * N_TOT + r] = f2bf(h0);
        hT[(size_t)nb * N_TOT + r] = f2bf(h1);
    }
}

// ---------------------------------------------------------------------------
// Kernel 2: s1 = h@W_a1 + b_a1 ; s2 = h@W_a2 + b_a2   (one wave per row)
// ---------------------------------------------------------------------------
__global__ __launch_bounds__(256) void gat_score_kernel(
    const float* __restrict__ h,
    const float* __restrict__ Wa1, const float* __restrict__ ba1,
    const float* __restrict__ Wa2, const float* __restrict__ ba2,
    float* __restrict__ s1, float* __restrict__ s2)
{
    const int wid  = threadIdx.x >> 5;
    const int lane = threadIdx.x & 31;
    const int row  = blockIdx.x * 8 + wid;
    const float* hr = h + (size_t)row * OUT_C;

    float p1 = 0.f, p2 = 0.f;
    #pragma unroll
    for (int c = lane; c < OUT_C; c += 32) {
        float v = hr[c];
        p1 += v * Wa1[c];
        p2 += v * Wa2[c];
    }
    #pragma unroll
    for (int off = 16; off > 0; off >>= 1) {
        p1 += __shfl_down(p1, off, 32);
        p2 += __shfl_down(p2, off, 32);
    }
    if (lane == 0) {
        s1[row] = p1 + ba1[0];
        s2[row] = p2 + ba2[0];
    }
}

// ---------------------------------------------------------------------------
// Kernel 3: fused masked-softmax attention:  out = softmax(mask(e)) @ h
// Flash-style over key blocks of 32. Block = 256 thr, 32 query rows.
// adj tiles double-buffered in LDS via async global->LDS loads (ASYNCcnt).
// Wave w: cols n0=32w..32w+31, both 16-row halves -> 4 accumulators.
// ---------------------------------------------------------------------------
#define NSTEP (N_TOT / 32)

__global__ __launch_bounds__(256) void gat_attn_kernel(
    const int* __restrict__ adj, const unsigned short* __restrict__ hT,
    const float* __restrict__ s1, const float* __restrict__ s2,
    float* __restrict__ out)
{
    __shared__ __align__(16) int            adjbuf[2][32][32]; // async dbl buffer
    __shared__ __align__(16) float          t[32][36];         // padded: no bank conflicts
    __shared__ __align__(16) unsigned short pmat[32][32];      // exp() tile, bf16
    __shared__ float mrow[32], lrow[32], srow[32];

    const int i0   = blockIdx.x * 32;
    const int tid  = threadIdx.x;
    const int wid  = tid >> 5;
    const int lane = tid & 31;
    const int hi   = lane >> 4;
    const int m    = lane & 15;
    const int n0   = wid * 32;

    v8f zero = {0.f,0.f,0.f,0.f,0.f,0.f,0.f,0.f};
    v8f acc00 = zero, acc01 = zero, acc10 = zero, acc11 = zero; // [it][nt]

    const int r4 = tid >> 3;          // copy/step-1 row (4 ints = 16B per thread)
    const int c4 = (tid & 7) * 4;
    const int* gadj = adj + (size_t)(i0 + r4) * N_TOT + c4;

    if (tid < 32) { mrow[tid] = -__builtin_inff(); lrow[tid] = 0.f; }

    // prologue: async-fetch tile 0 into buffer 0
    async_copy_b128(&adjbuf[0][r4][c4], gadj);
    __syncthreads();

    for (int js = 0; js < NSTEP; ++js) {
        const int j0  = js * 32;
        const int cur = js & 1;

        // ---- prefetch next adj tile into the other buffer, wait for current
        if (js + 1 < NSTEP) {
            async_copy_b128(&adjbuf[cur ^ 1][r4][c4], gadj + (size_t)(j0 + 32));
            asm volatile("s_wait_asynccnt 0x1" ::: "memory");  // tile js landed (in-order)
        } else {
            asm volatile("s_wait_asynccnt 0x0" ::: "memory");
        }
        __syncthreads();

        // ---- step 1: masked leaky-relu scores into LDS
        {
            int4 av = *(const int4*)&adjbuf[cur][r4][c4];
            float4 sv2 = *(const float4*)(s2 + j0 + c4);
            float sv1 = s1[i0 + r4];
            float e0 = sv1 + sv2.x; e0 = e0 > 0.f ? e0 : 0.01f * e0;
            float e1 = sv1 + sv2.y; e1 = e1 > 0.f ? e1 : 0.01f * e1;
            float e2 = sv1 + sv2.z; e2 = e2 > 0.f ? e2 : 0.01f * e2;
            float e3 = sv1 + sv2.w; e3 = e3 > 0.f ? e3 : 0.01f * e3;
            float4 tv;
            tv.x = av.x > 0 ? e0 : NEG_INF;
            tv.y = av.y > 0 ? e1 : NEG_INF;
            tv.z = av.z > 0 ? e2 : NEG_INF;
            tv.w = av.w > 0 ? e3 : NEG_INF;
            *(float4*)&t[r4][c4] = tv;
        }
        __syncthreads();

        // ---- step 2: online softmax stats per row (32 lanes, one row each)
        if (tid < 32) {
            const int r = tid;
            float mold = mrow[r];
            float tmax = -3.4e38f;
            #pragma unroll
            for (int c = 0; c < 32; ++c) tmax = fmaxf(tmax, t[r][c]);
            float mnew = fmaxf(mold, tmax);
            float sc = __expf(mold - mnew);           // -inf start -> 0
            float lsum = 0.f;
            #pragma unroll
            for (int c = 0; c < 32; ++c) {
                float pe = __expf(t[r][c] - mnew);
                pmat[r][c] = f2bf(pe);
                lsum += pe;
            }
            lrow[r] = lrow[r] * sc + lsum;
            mrow[r] = mnew;
            srow[r] = sc;
        }
        __syncthreads();

        // ---- step 3: rescale accumulators, then acc += p @ hT-block (WMMA)
        #pragma unroll
        for (int v = 0; v < 8; ++v) {
            float sc0 = srow[v + 8 * hi];             // rows 0..15  (it = 0)
            float sc1 = srow[16 + v + 8 * hi];        // rows 16..31 (it = 1)
            acc00[v] *= sc0; acc01[v] *= sc0;
            acc10[v] *= sc1; acc11[v] *= sc1;
        }

        // A frags (16x32 bf16 layout): lanes<16 K{0..15}, lanes>=16 K{16..31}
        const int klo = hi ? 16 : 0;
        const int khi = hi ? 24 : 8;
        v16bf afr0, afr1;
        {
            union { uint4 q[2]; v16bf v; } au;
            au.q[0] = *(const uint4*)&pmat[m][klo];
            au.q[1] = *(const uint4*)&pmat[m][khi];
            afr0 = au.v;
            au.q[0] = *(const uint4*)&pmat[16 + m][klo];
            au.q[1] = *(const uint4*)&pmat[16 + m][khi];
            afr1 = au.v;
        }

        // B frags: column n contiguous in K thanks to transposed hT (L2-resident)
        #pragma unroll
        for (int nt = 0; nt < 2; ++nt) {
            const int n = n0 + nt * 16 + m;
            const uint4* bp = (const uint4*)(hT + (size_t)n * N_TOT + j0 + 16 * hi);
            union { uint4 q[2]; v16bf v; } bu;
            bu.q[0] = bp[0];
            bu.q[1] = bp[1];
            if (nt == 0) {
                acc00 = __builtin_amdgcn_wmma_f32_16x16x32_bf16(false, afr0, false, bu.v, (short)0, acc00, false, false);
                acc10 = __builtin_amdgcn_wmma_f32_16x16x32_bf16(false, afr1, false, bu.v, (short)0, acc10, false, false);
            } else {
                acc01 = __builtin_amdgcn_wmma_f32_16x16x32_bf16(false, afr0, false, bu.v, (short)0, acc01, false, false);
                acc11 = __builtin_amdgcn_wmma_f32_16x16x32_bf16(false, afr1, false, bu.v, (short)0, acc11, false, false);
            }
        }
        __syncthreads();
    }

    // ---- finalize: divide by softmax denominator, write out
    #pragma unroll
    for (int v = 0; v < 8; ++v) {
        const int r0 = v + 8 * hi;
        const int r1 = 16 + v + 8 * hi;
        const float inv0 = 1.0f / lrow[r0];
        const float inv1 = 1.0f / lrow[r1];
        out[(size_t)(i0 + r0) * OUT_C + n0 + m]      = acc00[v] * inv0;
        out[(size_t)(i0 + r0) * OUT_C + n0 + 16 + m] = acc01[v] * inv0;
        out[(size_t)(i0 + r1) * OUT_C + n0 + m]      = acc10[v] * inv1;
        out[(size_t)(i0 + r1) * OUT_C + n0 + 16 + m] = acc11[v] * inv1;
    }
}

// ---------------------------------------------------------------------------
extern "C" void kernel_launch(void* const* d_in, const int* in_sizes, int n_in,
                              void* d_out, int out_size, void* d_ws, size_t ws_size,
                              hipStream_t stream) {
    const float* x   = (const float*)d_in[0];
    const int*   adj = (const int*)  d_in[1];
    const float* Wfc = (const float*)d_in[2];
    const float* bfc = (const float*)d_in[3];
    const float* Wa1 = (const float*)d_in[4];
    const float* ba1 = (const float*)d_in[5];
    const float* Wa2 = (const float*)d_in[6];
    const float* ba2 = (const float*)d_in[7];
    float* out = (float*)d_out;

    // workspace layout: h f32 (8 MB) | hT bf16 (4 MB) | s1 (32 KB) | s2 (32 KB)
    char* ws = (char*)d_ws;
    float*          h  = (float*)ws;
    unsigned short* hT = (unsigned short*)(ws + (size_t)N_TOT * OUT_C * sizeof(float));
    float*          s1 = (float*)(ws + (size_t)N_TOT * OUT_C * sizeof(float)
                                     + (size_t)OUT_C * N_TOT * sizeof(unsigned short));
    float*          s2 = s1 + N_TOT;

    gat_fc_kernel   <<<N_TOT / 16, 256, 0, stream>>>(x, Wfc, bfc, h, hT);
    gat_score_kernel<<<N_TOT / 8,  256, 0, stream>>>(h, Wa1, ba1, Wa2, ba2, s1, s2);
    gat_attn_kernel <<<N_TOT / 32, 256, 0, stream>>>(adj, hT, s1, s2, out);
}
